// RoPEAttention_60284160967333
// MI455X (gfx1250) — matrix operand order repaired and checked by reading, not compiled, not run on hardware
//
#include <hip/hip_runtime.h>
#include <hip/hip_bf16.h>

typedef __bf16 bf16_t;
typedef __attribute__((ext_vector_type(16))) __bf16 v16bf;
typedef __attribute__((ext_vector_type(8)))  __bf16 v8bf;
typedef __attribute__((ext_vector_type(8)))  float  v8f;
typedef int v4i_gcc __attribute__((vector_size(4 * sizeof(int))));

#define B_  2
#define L_  2048
#define D_  1024
#define H_  16
#define HD_ 64
#define M_  (B_ * L_)   // 4096 rows

// ---- CDNA5 async global->LDS path (guarded; falls back to sync staging) ----
#if defined(__has_builtin)
#  if __has_builtin(__builtin_amdgcn_global_load_async_to_lds_b128) && \
      __has_builtin(__builtin_amdgcn_s_wait_asynccnt)
#    define USE_ASYNC_LDS 1
#  endif
#endif
#ifndef USE_ASYNC_LDS
#  define USE_ASYNC_LDS 0
#endif

static __device__ inline void cp16_to_lds(const bf16_t* g, bf16_t* l) {
#if USE_ASYNC_LDS
  __builtin_amdgcn_global_load_async_to_lds_b128(
      (__attribute__((address_space(1))) v4i_gcc*)(void*)g,
      (__attribute__((address_space(3))) v4i_gcc*)(void*)l, 0, 0);
#else
  *(uint4*)l = *(const uint4*)g;
#endif
}

static __device__ inline void wait_async_lds() {
#if USE_ASYNC_LDS
  __builtin_amdgcn_s_wait_asynccnt(0);
#endif
}

static __device__ inline v8f wmma_bf16(v16bf a, v16bf b, v8f c) {
  return __builtin_amdgcn_wmma_f32_16x16x32_bf16(false, a, false, b, (short)0, c,
                                                 false, false);
}

// ---------------------------------------------------------------- convert
__global__ void f32_to_bf16_kernel(const float* __restrict__ in,
                                   bf16_t* __restrict__ out, int n) {
  int i = blockIdx.x * blockDim.x + threadIdx.x;
  if (i < n) out[i] = (bf16_t)in[i];
}

// ---------------------------------------------------------------- GEMM (+RoPE)
// C[M,N] = A[M,K] @ Bw[K,N] + bias ; optional interleaved-pair RoPE on 64-wide
// heads (pos = row % L_), optional bf16 vs f32 output.
__global__ __launch_bounds__(256, 2)
void gemm_bf16_rope(const bf16_t* __restrict__ A, const bf16_t* __restrict__ Bw,
                    const float* __restrict__ bias, void* Cout,
                    int M, int N, int K, int do_rope, int out_bf16) {
  constexpr int BM = 128, BN = 128, BK = 32, PAD = 8;
  __shared__ __align__(16) bf16_t As[BM][BK + PAD];
  __shared__ __align__(16) bf16_t Bs[BN][BK + PAD];

  const int tid  = threadIdx.x;
  const int lane = tid & 31;
  const int wv   = tid >> 5;
  const int half = lane >> 4;
  const int l15  = lane & 15;
  const int m0 = blockIdx.y * BM;
  const int n0 = blockIdx.x * BN;
  const int waveM = wv >> 1;  // 0..3
  const int waveN = wv & 1;   // 0..1

  v8f acc[2][4];
#pragma unroll
  for (int mf = 0; mf < 2; ++mf)
#pragma unroll
    for (int nf = 0; nf < 4; ++nf) acc[mf][nf] = (v8f){};

  const int arow = tid >> 1, ako = (tid & 1) * 16;  // A-tile stage role
  const int bkr  = tid & 31, bcs = (tid >> 5) * 16; // B-tile stage role
  const bf16_t* aptr = A + (size_t)(m0 + arow) * K + ako;
  const bf16_t* bptr = Bw + (size_t)bkr * N + n0 + bcs;

  for (int kk = 0; kk < K; kk += BK) {
    __syncthreads();
    // stage A (row-major, K contiguous) -- async DMA into LDS on CDNA5
    {
      const bf16_t* src = aptr + kk;
      cp16_to_lds(src,     &As[arow][ako]);
      cp16_to_lds(src + 8, &As[arow][ako + 8]);
    }
    // stage B transposed: Bs[n][k] (transpose -> keep sync path)
    {
      const bf16_t* src = bptr + (size_t)kk * N;
      bf16_t tmp[16];
      *(uint4*)&tmp[0] = *(const uint4*)src;
      *(uint4*)&tmp[8] = *(const uint4*)(src + 8);
#pragma unroll
      for (int j = 0; j < 16; ++j) Bs[bcs + j][bkr] = tmp[j];
    }
    // warm L2 for next iteration's tiles
    if (kk + BK < K) {
      __builtin_prefetch(aptr + kk + BK, 0, 1);
      __builtin_prefetch(bptr + (size_t)(kk + BK) * N, 0, 1);
    }
    wait_async_lds();
    __syncthreads();

    v16bf af[2], bfr[4];
#pragma unroll
    for (int mf = 0; mf < 2; ++mf) {
      const int row = waveM * 32 + mf * 16 + l15;
      ((v8bf*)&af[mf])[0] = *(const v8bf*)&As[row][half * 8];
      ((v8bf*)&af[mf])[1] = *(const v8bf*)&As[row][16 + half * 8];
    }
#pragma unroll
    for (int nf = 0; nf < 4; ++nf) {
      const int col = waveN * 64 + nf * 16 + l15;
      ((v8bf*)&bfr[nf])[0] = *(const v8bf*)&Bs[col][half * 8];
      ((v8bf*)&bfr[nf])[1] = *(const v8bf*)&Bs[col][16 + half * 8];
    }
#pragma unroll
    for (int mf = 0; mf < 2; ++mf)
#pragma unroll
      for (int nf = 0; nf < 4; ++nf)
        acc[mf][nf] = wmma_bf16(af[mf], bfr[nf], acc[mf][nf]);
  }

  // epilogue: bias (+RoPE) + store
#pragma unroll
  for (int mf = 0; mf < 2; ++mf) {
#pragma unroll
    for (int nf = 0; nf < 4; ++nf) {
      const int col  = n0 + waveN * 64 + nf * 16 + l15;
      const float bc = bias[col];
      const int hd   = col & (HD_ - 1);
      const int i2   = hd & ~1;
      const float invf = __expf(-(float)i2 * (9.2103403720f / 64.0f)); // 10000^(-2i/64)
#pragma unroll
      for (int r = 0; r < 8; ++r) {
        const int row = m0 + waveM * 32 + mf * 16 + r + half * 8;
        float v = acc[mf][nf][r] + bc;
        if (do_rope) {
          const float th = (float)(row & (L_ - 1)) * invf;
          float sn, cs;
          __sincosf(th, &sn, &cs);
          const float part = __shfl_xor(v, 1, 32); // rotary partner (adjacent N)
          v = (hd & 1) ? fmaf(v, cs, part * sn) : fmaf(v, cs, -part * sn);
        }
        const size_t idx = (size_t)row * N + col;
        if (out_bf16)
          ((bf16_t*)Cout)[idx] = (bf16_t)v;
        else
          ((float*)Cout)[idx] = v;
      }
    }
  }
}

// ---------------------------------------------------------------- flash attention
// Q,K,V token-major bf16 (B*L, D); each wave owns 16 q rows; 64-key LDS tiles,
// double-buffered with async global->LDS DMA.
__global__ __launch_bounds__(256, 2)
void flash_attn_kernel(const bf16_t* __restrict__ Q, const bf16_t* __restrict__ Kt,
                       const bf16_t* __restrict__ Vt, bf16_t* __restrict__ Out) {
  constexpr int KB = 64, PAD = 8, W = HD_ + PAD, NT = L_ / KB;
  __shared__ __align__(16) bf16_t Ks[2][KB][W];
  __shared__ __align__(16) bf16_t Vs[2][KB][W];
  __shared__ __align__(16) bf16_t Ps[8][16][W];  // per-wave P scratch (16 x 64)

  const int tid  = threadIdx.x;
  const int lane = tid & 31;
  const int wv   = tid >> 5;
  const int half = lane >> 4;
  const int l15  = lane & 15;
  const int bh = blockIdx.y;
  const int b  = bh / H_;
  const int h  = bh % H_;
  const size_t base = (size_t)b * L_ * D_ + (size_t)h * HD_;
  const int q0 = blockIdx.x * 128 + wv * 16;

  // persistent Q fragments (16 x 64 = two 16x32 A-frags)
  v16bf qf[2];
  {
    const bf16_t* qp = Q + base + (size_t)(q0 + l15) * D_;
#pragma unroll
    for (int c = 0; c < 2; ++c) {
      ((v8bf*)&qf[c])[0] = *(const v8bf*)(qp + c * 32 + half * 8);
      ((v8bf*)&qf[c])[1] = *(const v8bf*)(qp + c * 32 + 16 + half * 8);
    }
  }

  v8f o[4];
#pragma unroll
  for (int n = 0; n < 4; ++n) o[n] = (v8f){};
  float mrow[8], lrow[8];
#pragma unroll
  for (int r = 0; r < 8; ++r) { mrow[r] = -1e30f; lrow[r] = 0.f; }

  const int ldrow = tid >> 2, ldseg = (tid & 3) * 16;
  const bf16_t* kbase = Kt + base + (size_t)ldrow * D_ + ldseg;
  const bf16_t* vbase = Vt + base + (size_t)ldrow * D_ + ldseg;

  // stage tile kb into buffer buf
  auto stage = [&](int kb, int buf) {
    const bf16_t* ks = kbase + (size_t)kb * KB * D_;
    const bf16_t* vs = vbase + (size_t)kb * KB * D_;
    cp16_to_lds(ks,     &Ks[buf][ldrow][ldseg]);
    cp16_to_lds(ks + 8, &Ks[buf][ldrow][ldseg + 8]);
    cp16_to_lds(vs,     &Vs[buf][ldrow][ldseg]);
    cp16_to_lds(vs + 8, &Vs[buf][ldrow][ldseg + 8]);
  };

  stage(0, 0);

  for (int kb = 0; kb < NT; ++kb) {
    const int buf = kb & 1;
    wait_async_lds();
    __syncthreads();                       // tile `buf` visible to all waves
    if (kb + 1 < NT) stage(kb + 1, buf ^ 1);  // overlap DMA with compute

    // S = Q @ K^T * 1/sqrt(HD) : four 16x16 tiles over 64 keys
    v8f s4[4];
#pragma unroll
    for (int t = 0; t < 4; ++t) {
      v8f s = (v8f){};
      const int key = t * 16 + l15;
#pragma unroll
      for (int c = 0; c < 2; ++c) {
        v16bf kf;
        ((v8bf*)&kf)[0] = *(const v8bf*)&Ks[buf][key][c * 32 + half * 8];
        ((v8bf*)&kf)[1] = *(const v8bf*)&Ks[buf][key][c * 32 + 16 + half * 8];
        s = wmma_bf16(qf[c], kf, s);
      }
      s4[t] = s * 0.125f;
    }

    // online softmax (rows split lanes0-15 = r, lanes16-31 = r+8)
#pragma unroll
    for (int r = 0; r < 8; ++r) {
      float mx = fmaxf(fmaxf(s4[0][r], s4[1][r]), fmaxf(s4[2][r], s4[3][r]));
      mx = fmaxf(mx, __shfl_xor(mx, 8, 32));
      mx = fmaxf(mx, __shfl_xor(mx, 4, 32));
      mx = fmaxf(mx, __shfl_xor(mx, 2, 32));
      mx = fmaxf(mx, __shfl_xor(mx, 1, 32));
      const float mn   = fmaxf(mrow[r], mx);
      const float corr = __expf(mrow[r] - mn);
      mrow[r] = mn;
      lrow[r] *= corr;
      o[0][r] *= corr; o[1][r] *= corr; o[2][r] *= corr; o[3][r] *= corr;
      float part = 0.f;
#pragma unroll
      for (int t = 0; t < 4; ++t) {
        const float p = __expf(s4[t][r] - mn);
        s4[t][r] = p;
        part += p;
      }
      part += __shfl_xor(part, 8, 32);
      part += __shfl_xor(part, 4, 32);
      part += __shfl_xor(part, 2, 32);
      part += __shfl_xor(part, 1, 32);
      lrow[r] += part;
    }

    // C-layout -> A-layout via per-wave LDS bounce
#pragma unroll
    for (int t = 0; t < 4; ++t)
#pragma unroll
      for (int r = 0; r < 8; ++r)
        Ps[wv][r + half * 8][t * 16 + l15] = (bf16_t)s4[t][r];

    // O += P @ V
#pragma unroll
    for (int c = 0; c < 2; ++c) {
      v16bf pf;
      ((v8bf*)&pf)[0] = *(const v8bf*)&Ps[wv][l15][c * 32 + half * 8];
      ((v8bf*)&pf)[1] = *(const v8bf*)&Ps[wv][l15][c * 32 + 16 + half * 8];
#pragma unroll
      for (int n = 0; n < 4; ++n) {
        v16bf vf;
#pragma unroll
        for (int e = 0; e < 16; ++e)
          vf[e] = Vs[buf][c * 32 + half * 8 + e + (e >= 8 ? 8 : 0)][n * 16 + l15];
        o[n] = wmma_bf16(pf, vf, o[n]);
      }
    }
  }

  // normalize and store token-major bf16 (feeds output projection)
#pragma unroll
  for (int n = 0; n < 4; ++n) {
#pragma unroll
    for (int r = 0; r < 8; ++r) {
      const float val = o[n][r] / lrow[r];
      const int row = q0 + r + half * 8;
      Out[(size_t)(b * L_ + row) * D_ + h * HD_ + n * 16 + l15] = (bf16_t)val;
    }
  }
}

// ---------------------------------------------------------------- launch
extern "C" void kernel_launch(void* const* d_in, const int* in_sizes, int n_in,
                              void* d_out, int out_size, void* d_ws, size_t ws_size,
                              hipStream_t stream) {
  (void)in_sizes; (void)n_in; (void)out_size; (void)ws_size;
  const float* x  = (const float*)d_in[0];
  const float* Wq = (const float*)d_in[1];
  const float* bq = (const float*)d_in[2];
  const float* Wk = (const float*)d_in[3];
  const float* bk = (const float*)d_in[4];
  const float* Wv = (const float*)d_in[5];
  const float* bv = (const float*)d_in[6];
  const float* Wp = (const float*)d_in[7];
  const float* bp = (const float*)d_in[8];

  char* ws = (char*)d_ws;
  size_t off = 0;
  bf16_t* xb  = (bf16_t*)(ws + off); off += (size_t)M_ * D_ * 2;
  bf16_t* Wqb = (bf16_t*)(ws + off); off += (size_t)D_ * D_ * 2;
  bf16_t* Wkb = (bf16_t*)(ws + off); off += (size_t)D_ * D_ * 2;
  bf16_t* Wvb = (bf16_t*)(ws + off); off += (size_t)D_ * D_ * 2;
  bf16_t* Wpb = (bf16_t*)(ws + off); off += (size_t)D_ * D_ * 2;
  bf16_t* Qb  = (bf16_t*)(ws + off); off += (size_t)M_ * D_ * 2;
  bf16_t* Kb  = (bf16_t*)(ws + off); off += (size_t)M_ * D_ * 2;
  bf16_t* Vb  = (bf16_t*)(ws + off); off += (size_t)M_ * D_ * 2;
  bf16_t* Ab  = (bf16_t*)(ws + off); off += (size_t)M_ * D_ * 2;

  const int nx = M_ * D_, nw = D_ * D_;
  f32_to_bf16_kernel<<<(nx + 255) / 256, 256, 0, stream>>>(x,  xb,  nx);
  f32_to_bf16_kernel<<<(nw + 255) / 256, 256, 0, stream>>>(Wq, Wqb, nw);
  f32_to_bf16_kernel<<<(nw + 255) / 256, 256, 0, stream>>>(Wk, Wkb, nw);
  f32_to_bf16_kernel<<<(nw + 255) / 256, 256, 0, stream>>>(Wv, Wvb, nw);
  f32_to_bf16_kernel<<<(nw + 255) / 256, 256, 0, stream>>>(Wp, Wpb, nw);

  dim3 gg(D_ / 128, M_ / 128);  // (8, 32)
  gemm_bf16_rope<<<gg, 256, 0, stream>>>(xb, Wqb, bq, Qb, M_, D_, D_, 1, 1);
  gemm_bf16_rope<<<gg, 256, 0, stream>>>(xb, Wkb, bk, Kb, M_, D_, D_, 1, 1);
  gemm_bf16_rope<<<gg, 256, 0, stream>>>(xb, Wvb, bv, Vb, M_, D_, D_, 0, 1);

  flash_attn_kernel<<<dim3(L_ / 128, B_ * H_), 256, 0, stream>>>(Qb, Kb, Vb, Ab);

  gemm_bf16_rope<<<gg, 256, 0, stream>>>(Ab, Wpb, bp, (float*)d_out,
                                         M_, D_, D_, 0, 0);
}
